// AttDGCNN_65000035058610
// MI455X (gfx1250) — matrix operand order repaired and checked by reading, not compiled, not run on hardware
//
#include <hip/hip_runtime.h>
#include <hip/hip_bf16.h>
#include <stdint.h>

// ---------------------------------------------------------------------------
// AttDGCNN forward for gfx1250 (MI455X).
//  - Edge-MLP / residual / head GEMMs: v_wmma_f32_16x16x32_bf16, block-tiled:
//    8 waves x (16 x Nblk) tiles; weight strip staged in LDS via the Tensor
//    Data Mover (tensor_load_to_lds + s_wait_tensorcnt), B frags via ds_load.
//  - kNN pairwise-distance Gram matrix: v_wmma_f32_16x16x4_f32 (full f32).
//  - Workspace requirement ~350 MB (carved below).
// ---------------------------------------------------------------------------

typedef __attribute__((ext_vector_type(16))) __bf16 v16bf;
typedef __attribute__((ext_vector_type(8)))  float  v8f;
typedef __attribute__((ext_vector_type(2)))  float  v2f;
typedef __attribute__((ext_vector_type(4)))  unsigned int v4u;
typedef __attribute__((ext_vector_type(8)))  int    v8i;
typedef __attribute__((ext_vector_type(4)))  int    v4i;

#define BATCH 8
#define NPTS  1024
#define BN    (BATCH*NPTS)
#define KNN   30
#define ME    (BN*KNN)
#define FEATD 512

#if defined(__has_builtin)
#if __has_builtin(__builtin_amdgcn_wmma_f32_16x16x4_f32)
#define HAVE_WMMA_F32 1
#endif
#if __has_builtin(__builtin_amdgcn_tensor_load_to_lds)
#define HAVE_TDM 1
#endif
#endif

static __device__ __forceinline__ void wait_tensorcnt0() {
#if defined(__has_builtin)
#if __has_builtin(__builtin_amdgcn_s_wait_tensorcnt)
  __builtin_amdgcn_s_wait_tensorcnt(0);
  return;
#else
  asm volatile("s_wait_tensorcnt 0x0" ::: "memory");
  return;
#endif
#else
  asm volatile("s_wait_tensorcnt 0x0" ::: "memory");
#endif
}

// ------------------------- small prep / elementwise -------------------------

__global__ void k_cvt_x0(const float* __restrict__ src, float* __restrict__ xf,
                         __bf16* __restrict__ xb) {
  int t = blockIdx.x * blockDim.x + threadIdx.x;       // over BN*32
  if (t >= BN * 32) return;
  int c = t & 31, row = t >> 5;
  float v = (c < 3) ? src[row * 3 + c] : 0.f;
  if (c < 4) xf[row * 4 + c] = v;
  xb[row * 32 + c] = (__bf16)v;
}

__global__ void k_sq(const float* __restrict__ xf, float* __restrict__ sq, int C, int xs) {
  int r = blockIdx.x * blockDim.x + threadIdx.x;
  if (r >= BN) return;
  const float* p = xf + (size_t)r * xs;
  float s = 0.f;
  for (int c = 0; c < C; ++c) s += p[c] * p[c];
  sq[r] = s;
}

// W f32 [Kdim rows x (rsw cols)], take cols [co, co+Nout) -> Wt bf16 [Npad][Kpad], 0-padded
__global__ void k_prepw(const float* __restrict__ W, __bf16* __restrict__ Wt,
                        int Kdim, int Nout, int rsw, int co, int Kpad, int Npad, long total) {
  long t = (long)blockIdx.x * blockDim.x + threadIdx.x;
  if (t >= total) return;
  int kk = (int)(t % Kpad);
  int n  = (int)(t / Kpad);
  float v = (n < Nout && kk < Kdim) ? W[(size_t)kk * rsw + co + n] : 0.f;
  Wt[(size_t)n * Kpad + kk] = (__bf16)v;
}

__global__ void k_prepb(const float* __restrict__ b, float* __restrict__ bp, int Nout, int Npad) {
  int t = blockIdx.x * blockDim.x + threadIdx.x;
  if (t >= Npad) return;
  bp[t] = (t < Nout) ? b[t] : 0.f;
}

// ------------------------- kNN: distance (f32 WMMA) + top-K -----------------

__global__ void k_dist(const float* __restrict__ xf, const float* __restrict__ sq,
                       float* __restrict__ dist, int Cpad, int xs) {
  int wave = (blockIdx.x * blockDim.x + threadIdx.x) >> 5;
  int lane = threadIdx.x & 31;
  const int TN = NPTS / 16;        // 64 tiles per edge
  const int TPB = TN * TN;         // 4096 tiles per batch
  int b = wave / TPB;
  if (b >= BATCH) return;
  int t = wave % TPB;
  int tm = t / TN, tn = t % TN;
  const float* Xb = xf + (size_t)b * NPTS * xs;
  int hi = lane >> 4, l15 = lane & 15;
  int m = tm * 16 + l15;
  int n = tn * 16 + l15;
  int koff = hi * 2;
  union { v8f v; float f[8]; } acc;
  acc.v = (v8f){};
  int mbase = tm * 16 + hi * 8;
#if HAVE_WMMA_F32
  for (int k = 0; k < Cpad; k += 4) {
    v2f a, bb;
    a.x  = Xb[(size_t)m * xs + k + koff];
    a.y  = Xb[(size_t)m * xs + k + koff + 1];
    bb.x = Xb[(size_t)n * xs + k + koff];
    bb.y = Xb[(size_t)n * xs + k + koff + 1];
    acc.v = __builtin_amdgcn_wmma_f32_16x16x4_f32(false, a, false, bb, (short)0, acc.v,
                                                  false, false);
  }
#else
  for (int r = 0; r < 8; ++r) {
    float s = 0.f;
    for (int k = 0; k < Cpad; ++k)
      s += Xb[(size_t)(mbase + r) * xs + k] * Xb[(size_t)n * xs + k];
    acc.f[r] = s;
  }
#endif
  const float* sqb = sq + b * NPTS;
  float* Db = dist + (size_t)b * NPTS * NPTS;
#pragma unroll
  for (int r = 0; r < 8; ++r) {
    int mm = mbase + r;
    Db[(size_t)mm * NPTS + n] = sqb[mm] - 2.f * acc.f[r] + sqb[n];
  }
}

__global__ void k_topk(const float* __restrict__ dist, int* __restrict__ idx) {
  int r = blockIdx.x * blockDim.x + threadIdx.x;
  if (r >= BN) return;
  int b = r / NPTS;
  const float* row = dist + (size_t)b * NPTS * NPTS + (size_t)(r % NPTS) * NPTS;
  float bd[KNN];
  int bi[KNN];
  for (int i = 0; i < KNN; ++i) { bd[i] = 3.4e38f; bi[i] = 0; }
  for (int j = 0; j < NPTS; ++j) {
    float d = row[j];
    if (d < bd[KNN - 1]) {
      int p = KNN - 1;
      while (p > 0 && bd[p - 1] > d) { bd[p] = bd[p - 1]; bi[p] = bi[p - 1]; --p; }
      bd[p] = d; bi[p] = j;
    }
  }
  for (int i = 0; i < KNN; ++i) idx[(size_t)r * KNN + i] = bi[i];
}

// edge features e = [xi, xj-xi] -> bf16, K padded
__global__ void k_gather(const float* __restrict__ xf, const int* __restrict__ idx,
                         __bf16* __restrict__ E, int Cin, int xs, int Kpad, long total) {
  long t = (long)blockIdx.x * blockDim.x + threadIdx.x;
  if (t >= total) return;
  int c = (int)(t % Kpad);
  long row = t / Kpad;              // edge row = (b*1024+n)*30 + k
  long pn = row / KNN;              // b*1024+n
  int b = (int)(pn / NPTS);
  int j = idx[row];
  float v = 0.f;
  if (c < Cin) v = xf[pn * xs + c];
  else if (c < 2 * Cin) {
    int cc = c - Cin;
    v = xf[((long)b * NPTS + j) * xs + cc] - xf[pn * xs + cc];
  }
  E[row * Kpad + c] = (__bf16)v;
}

// ------------------------- block-tiled bf16 WMMA GEMM -----------------------
// C[M,Npad] = act(A[M,Kdim] @ Bt[Npad,Kdim]^T + bias[Npad]); Kdim%32==0.
// Block: 8 waves, tile 128 x (NT*16). Weight strip [Nblk][Kdim] staged into
// LDS once per block via the Tensor Data Mover, then read with ds_load_b128.
template <int NT>
__global__ void k_gemm_blk(const __bf16* __restrict__ A, const __bf16* __restrict__ Bt,
                           const float* __restrict__ bias, float* __restrict__ Cf,
                           __bf16* __restrict__ Cb, int M, int Npad, int Kdim, int relu) {
  extern __shared__ char smem_raw[];
  __bf16* lds_b = (__bf16*)smem_raw;

  const int Nblk = NT * 16;
  const int nb = blockIdx.y;
  const int wid = threadIdx.x >> 5;
  const int lane = threadIdx.x & 31;
  const int hi = lane >> 4, l15 = lane & 15;

  // ---- stage B strip [Nblk rows x Kdim] into LDS ----
#if HAVE_TDM
  if (threadIdx.x < 32) {   // single wave issues the TDM descriptor
    unsigned long long gaddr =
        (unsigned long long)(const char*)(Bt + (size_t)nb * Nblk * Kdim);
    v4u g0;
    g0.x = 1u;                                   // count=1, user descriptor
    g0.y = 0u;                                   // lds_addr = dynamic-LDS base (0)
    g0.z = (unsigned)gaddr;                      // global_addr[31:0]
    g0.w = (unsigned)(gaddr >> 32) | (2u << 30); // global_addr msbs | type=2
    v8i g1;
    g1[0] = (1 << 16);                                   // data_size = 2 bytes
    g1[1] = (int)((Kdim & 0xffff) << 16);                // tensor_dim0[15:0]
    g1[2] = (int)(((unsigned)Kdim >> 16) & 0xffff) |
            (int)((Nblk & 0xffff) << 16);                // dim0 hi | tensor_dim1 lo
    g1[3] = (int)((Kdim & 0xffff) << 16);                // dim1 hi(0) | tile_dim0 = Kdim
    g1[4] = Nblk & 0xffff;                               // tile_dim1 = Nblk, tile_dim2 = 0
    g1[5] = Kdim;                                        // tensor_dim0_stride lo
    g1[6] = 0;
    g1[7] = 0;
    v4i g2 = {0, 0, 0, 0};
    v4i g3 = {0, 0, 0, 0};
#if __clang_major__ >= 23
    v8i g4 = {0, 0, 0, 0, 0, 0, 0, 0};
    __builtin_amdgcn_tensor_load_to_lds(g0, g1, g2, g3, g4, 0);
#else
    __builtin_amdgcn_tensor_load_to_lds(g0, g1, g2, g3, 0);
#endif
    wait_tensorcnt0();
  }
#else
  {
    const uint4* src = (const uint4*)(Bt + (size_t)nb * Nblk * Kdim);
    uint4* dst = (uint4*)lds_b;
    int totv = Nblk * Kdim / 8;
    for (int i = threadIdx.x; i < totv; i += 256) dst[i] = src[i];
  }
#endif
  __syncthreads();

  int m0 = blockIdx.x * 128 + wid * 16;     // this wave's 16-row strip
  if (m0 >= M) return;

  // A frag: lane(0-15): M=l15, K 0..7 & 16..23 ; lane(16-31): K 8..15 & 24..31
  const __bf16* arow = A + (size_t)(m0 + l15) * Kdim + hi * 8;
  v8f acc[NT];
#pragma unroll
  for (int nt = 0; nt < NT; ++nt) acc[nt] = (v8f){};

  for (int k = 0; k < Kdim; k += 32) {
    union { v16bf v; uint4 u[2]; } a;
    a.u[0] = *(const uint4*)(arow + k);
    a.u[1] = *(const uint4*)(arow + k + 16);
#pragma unroll
    for (int nt = 0; nt < NT; ++nt) {
      // B frag: lane(0-15): N=l15, K 0..7 & 16..23 ; lane(16-31): K 8..15 & 24..31
      const __bf16* bl = lds_b + (size_t)(nt * 16 + l15) * Kdim + hi * 8 + k;
      union { v16bf v; uint4 u[2]; } bu;
      bu.u[0] = *(const uint4*)(bl);
      bu.u[1] = *(const uint4*)(bl + 16);
      acc[nt] = __builtin_amdgcn_wmma_f32_16x16x32_bf16(false, a.v, false, bu.v, (short)0,
                                                        acc[nt], false, false);
    }
  }

  int nbase = nb * Nblk;
  int mrow = m0 + hi * 8;
#pragma unroll
  for (int nt = 0; nt < NT; ++nt) {
    int n = nbase + nt * 16 + l15;
    float bv = bias ? bias[n] : 0.f;
    union { v8f v; float f[8]; } u;
    u.v = acc[nt];
#pragma unroll
    for (int r = 0; r < 8; ++r) {
      float v = u.f[r] + bv;
      if (relu) v = v > 0.f ? v : 0.f;
      size_t o = (size_t)(mrow + r) * Npad + n;
      if (Cf) Cf[o] = v;
      if (Cb) Cb[o] = (__bf16)v;
    }
  }
}

// ------------------------- aggregation / attention epilogues ----------------

__global__ void k_maxk(const __bf16* __restrict__ H3, float* __restrict__ h, int C, long total) {
  long t = (long)blockIdx.x * blockDim.x + threadIdx.x;
  if (t >= total) return;
  int c = (int)(t % C);
  long pn = t / C;
  const __bf16* p = H3 + (pn * KNN) * (size_t)C + c;
  float m = -3.4e38f;
  for (int k = 0; k < KNN; ++k) {
    float v = (float)p[(size_t)k * C];
    m = v > m ? v : m;
  }
  h[pn * (size_t)C + c] = m;
}

__global__ void k_meanp(const float* __restrict__ h, float* __restrict__ s0, int C) {
  int t = blockIdx.x * blockDim.x + threadIdx.x;
  if (t >= BATCH * C) return;
  int b = t / C, c = t % C;
  float s = 0.f;
  for (int n = 0; n < NPTS; ++n) s += h[((size_t)b * NPTS + n) * C + c];
  s0[t] = s / (float)NPTS;
}

__global__ void k_se1(const float* __restrict__ s0, const float* __restrict__ w1,
                      float* __restrict__ tmid, int C, int Cmid) {
  int t = blockIdx.x * blockDim.x + threadIdx.x;
  if (t >= BATCH * Cmid) return;
  int b = t / Cmid, j = t % Cmid;
  float s = 0.f;
  for (int c = 0; c < C; ++c) s += s0[b * C + c] * w1[c * Cmid + j];
  tmid[t] = s > 0.f ? s : 0.f;
}

__global__ void k_se2(const float* __restrict__ tmid, const float* __restrict__ w2,
                      float* __restrict__ s, int C, int Cmid) {
  int t = blockIdx.x * blockDim.x + threadIdx.x;
  if (t >= BATCH * C) return;
  int b = t / C, c = t % C;
  float a = 0.f;
  for (int j = 0; j < Cmid; ++j) a += tmid[b * Cmid + j] * w2[j * C + c];
  s[t] = 1.f / (1.f + expf(-a));
}

__global__ void k_seapply(float* h, const float* __restrict__ s, float* __restrict__ y2, int C) {
  int pn = blockIdx.x * blockDim.x + threadIdx.x;
  if (pn >= BN) return;
  int b = pn / NPTS;
  float mean = 0.f, mx = -3.4e38f;
  for (int c = 0; c < C; ++c) {
    float v = h[(size_t)pn * C + c] * s[b * C + c];
    h[(size_t)pn * C + c] = v;
    mean += v;
    mx = v > mx ? v : mx;
  }
  y2[pn * 2 + 0] = mean / (float)C;
  y2[pn * 2 + 1] = mx;
}

__global__ void k_spconv(const float* __restrict__ y2, const float* __restrict__ w,
                         float* __restrict__ y) {
  int pn = blockIdx.x * blockDim.x + threadIdx.x;
  if (pn >= BN) return;
  int n = pn % NPTS, b = pn / NPTS;
  float a = 0.f;
  for (int t = 0; t < 7; ++t) {
    int nn = n - 3 + t;
    if (nn >= 0 && nn < NPTS) {
      const float* p = y2 + ((size_t)b * NPTS + nn) * 2;
      a += p[0] * w[t * 2 + 0] + p[1] * w[t * 2 + 1];
    }
  }
  y[pn] = 1.f / (1.f + expf(-a));
}

__global__ void k_apply(const float* h, const float* __restrict__ y, const float* res,
                        float* xf, __bf16* __restrict__ xb, float* __restrict__ feats,
                        int C, int off, long total) {
  long t = (long)blockIdx.x * blockDim.x + threadIdx.x;
  if (t >= total) return;
  int c = (int)(t % C);
  long pn = t / C;
  float v = h[pn * C + c] * y[pn] + res[pn * C + c];
  xf[pn * C + c] = v;
  xb[pn * C + c] = (__bf16)v;
  feats[pn * FEATD + off + c] = v;
}

__global__ void k_pool(const float* __restrict__ feats, float* __restrict__ g) {
  int t = blockIdx.x * blockDim.x + threadIdx.x;
  if (t >= BATCH * FEATD) return;
  int b = t / FEATD, c = t % FEATD;
  float mx = -3.4e38f, sm = 0.f;
  for (int n = 0; n < NPTS; ++n) {
    float v = feats[((size_t)b * NPTS + n) * FEATD + c];
    mx = v > mx ? v : mx;
    sm += v;
  }
  g[t] = mx + sm / (float)NPTS;
}

__global__ void k_padg(const float* __restrict__ g, __bf16* __restrict__ gp) {
  int t = blockIdx.x * blockDim.x + threadIdx.x;
  if (t >= 16 * FEATD) return;
  int r = t / FEATD;
  gp[t] = (__bf16)((r < BATCH) ? g[t] : 0.f);
}

__global__ void k_copyout(const float* __restrict__ hf, float* __restrict__ out) {
  int t = blockIdx.x * blockDim.x + threadIdx.x;
  if (t >= BATCH * 49) return;
  int b = t / 49, j = t % 49;
  out[t] = hf[b * 64 + j];
}

// ------------------------------- host side ---------------------------------

static inline char* carve(char*& p, size_t bytes) {
  char* r = p;
  p += (bytes + 255) & ~(size_t)255;
  return r;
}

static void prep(const float* W, int Kdim, int Nout, int rsw, int co, int Kpad, int Npad,
                 __bf16* wt, const float* bsrc, float* bdst, hipStream_t s) {
  long tot = (long)Npad * Kpad;
  k_prepw<<<(unsigned)((tot + 255) / 256), 256, 0, s>>>(W, wt, Kdim, Nout, rsw, co, Kpad, Npad, tot);
  if (bsrc) k_prepb<<<(Npad + 255) / 256, 256, 0, s>>>(bsrc, bdst, Nout, Npad);
}

static void gemm(const __bf16* A, const __bf16* Bt, const float* bias, float* Cf, __bf16* Cb,
                 int M, int Npad, int Kdim, int relu, hipStream_t s) {
  int mblocks = (M + 127) / 128;
  if (Npad >= 128) {
    dim3 g(mblocks, Npad / 128);
    size_t sm = (size_t)128 * Kdim * 2;
    k_gemm_blk<8><<<g, 256, sm, s>>>(A, Bt, bias, Cf, Cb, M, Npad, Kdim, relu);
  } else {
    dim3 g(mblocks, Npad / 64);
    size_t sm = (size_t)64 * Kdim * 2;
    k_gemm_blk<4><<<g, 256, sm, s>>>(A, Bt, bias, Cf, Cb, M, Npad, Kdim, relu);
  }
}

extern "C" void kernel_launch(void* const* d_in, const int* in_sizes, int n_in,
                              void* d_out, int out_size, void* d_ws, size_t ws_size,
                              hipStream_t stream) {
  (void)in_sizes; (void)n_in; (void)out_size; (void)ws_size;

  // d_in flat order: data, then ec0..ec3 params, then head params.
  const int CIN[4] = {3, 64, 64, 128}, COUT[4] = {64, 64, 128, 256};
  const int IW1[4] = {1, 12, 21, 32}, IB1[4] = {2, 13, 22, 33};
  const int IW2[4] = {3, 14, 23, 34}, IB2[4] = {4, 15, 24, 35};
  const int IW3[4] = {5, 16, 25, 36}, IB3[4] = {6, 17, 26, 37};
  const int ISE1[4] = {7, 18, 27, 38}, ISE2[4] = {8, 19, 28, 39}, ISP[4] = {9, 20, 29, 40};
  const int IRW[4] = {10, -1, 30, 41}, IRB[4] = {11, -1, 31, 42};
  const int IQKV = 43, IPROJ = 44, IPROJB = 45, IPF1 = 46, IPF1B = 47, IPF2 = 48, IPF2B = 49,
            IO1 = 50, IO1B = 51, IO2 = 52, IO2B = 53, IO3 = 54, IO3B = 55;

  char* p = (char*)d_ws;
  float*  XF32  = (float*)carve(p, (size_t)BN * 256 * 4);
  __bf16* XB16  = (__bf16*)carve(p, (size_t)BN * 256 * 2);
  float*  SQ    = (float*)carve(p, (size_t)BN * 4);
  float*  DIST  = (float*)carve(p, (size_t)BATCH * NPTS * NPTS * 4);
  int*    IDX   = (int*)carve(p, (size_t)ME * 4);
  __bf16* EBUF  = (__bf16*)carve(p, (size_t)ME * 256 * 2);
  __bf16* HBUF  = (__bf16*)carve(p, (size_t)ME * 256 * 2);
  float*  H     = (float*)carve(p, (size_t)BN * 256 * 4);
  float*  RES   = (float*)carve(p, (size_t)BN * 256 * 4);
  float*  S0    = (float*)carve(p, 8 * 256 * 4);
  float*  TMID  = (float*)carve(p, 8 * 16 * 4);
  float*  S     = (float*)carve(p, 8 * 256 * 4);
  float*  Y2    = (float*)carve(p, (size_t)BN * 2 * 4);
  float*  Y     = (float*)carve(p, (size_t)BN * 4);
  float*  FEATS = (float*)carve(p, (size_t)BN * FEATD * 4);
  __bf16* WT    = (__bf16*)carve(p, (size_t)1024 * 1024 * 2);
  float*  BB    = (float*)carve(p, 1024 * 4);
  float*  G     = (float*)carve(p, 8 * 512 * 4);
  __bf16* GPAD  = (__bf16*)carve(p, 16 * 512 * 2);
  __bf16* HA    = (__bf16*)carve(p, 16 * 1024 * 2);
  __bf16* HB    = (__bf16*)carve(p, 16 * 1024 * 2);
  float*  HEADF = (float*)carve(p, 16 * 64 * 4);

  const float* data = (const float*)d_in[0];
  k_cvt_x0<<<(BN * 32 + 255) / 256, 256, 0, stream>>>(data, XF32, XB16);

  int off = 0;
  for (int i = 0; i < 4; ++i) {
    int Cin = CIN[i], C = COUT[i];
    int xs = (i == 0) ? 4 : Cin;       // f32 feature stride (padded, multiple of 4)
    int Kx = (i == 0) ? 32 : Cin;      // bf16 feature stride (multiple of 32)
    int Ke = (i == 0) ? 32 : 2 * Cin;  // edge-feature K (multiple of 32)

    k_sq<<<(BN + 255) / 256, 256, 0, stream>>>(XF32, SQ, Cin, xs);
    k_dist<<<BATCH * 4096 / 8, 256, 0, stream>>>(XF32, SQ, DIST, xs, xs);
    k_topk<<<(BN + 255) / 256, 256, 0, stream>>>(DIST, IDX);

    long totE = (long)ME * Ke;
    k_gather<<<(unsigned)((totE + 255) / 256), 256, 0, stream>>>(XF32, IDX, EBUF, Cin, xs, Ke, totE);

    if (IRW[i] >= 0) {  // residual 1x1 projection
      prep((const float*)d_in[IRW[i]], Cin, C, C, 0, Kx, C, WT, (const float*)d_in[IRB[i]], BB, stream);
      gemm(XB16, WT, BB, RES, nullptr, BN, C, Kx, 0, stream);
    }

    // edge MLP: relu(e@w1+b1) -> relu(@w2+b2) -> @w3+b3
    prep((const float*)d_in[IW1[i]], 2 * Cin, C, C, 0, Ke, C, WT, (const float*)d_in[IB1[i]], BB, stream);
    gemm(EBUF, WT, BB, nullptr, HBUF, ME, C, Ke, 1, stream);
    prep((const float*)d_in[IW2[i]], C, C, C, 0, C, C, WT, (const float*)d_in[IB2[i]], BB, stream);
    gemm(HBUF, WT, BB, nullptr, EBUF, ME, C, C, 1, stream);
    prep((const float*)d_in[IW3[i]], C, C, C, 0, C, C, WT, (const float*)d_in[IB3[i]], BB, stream);
    gemm(EBUF, WT, BB, nullptr, HBUF, ME, C, C, 0, stream);

    long tHC = (long)BN * C;
    k_maxk<<<(unsigned)((tHC + 255) / 256), 256, 0, stream>>>(HBUF, H, C, tHC);

    // SE channel attention
    int Cmid = C / 16;
    k_meanp<<<(BATCH * C + 255) / 256, 256, 0, stream>>>(H, S0, C);
    k_se1<<<(BATCH * Cmid + 255) / 256, 256, 0, stream>>>(S0, (const float*)d_in[ISE1[i]], TMID, C, Cmid);
    k_se2<<<(BATCH * C + 255) / 256, 256, 0, stream>>>(TMID, (const float*)d_in[ISE2[i]], S, C, Cmid);
    k_seapply<<<(BN + 255) / 256, 256, 0, stream>>>(H, S, Y2, C);

    // spatial attention + residual; also writes into feats slice
    k_spconv<<<(BN + 255) / 256, 256, 0, stream>>>(Y2, (const float*)d_in[ISP[i]], Y);
    const float* resptr = (IRW[i] >= 0) ? RES : XF32;
    k_apply<<<(unsigned)((tHC + 255) / 256), 256, 0, stream>>>(H, Y, resptr, XF32, XB16, FEATS, C, off, tHC);
    off += C;
  }

  // global max+mean pool
  k_pool<<<(BATCH * FEATD + 255) / 256, 256, 0, stream>>>(FEATS, G);
  k_padg<<<(16 * FEATD + 255) / 256, 256, 0, stream>>>(G, GPAD);

  // MHSA with seq-len 1: softmax(1x1)==1 => o = g @ qkv_w[:,1024:1536]
  prep((const float*)d_in[IQKV], 512, 512, 1536, 1024, 512, 512, WT, nullptr, BB, stream);
  gemm(GPAD, WT, nullptr, nullptr, HA, 16, 512, 512, 0, stream);
  prep((const float*)d_in[IPROJ], 512, 512, 512, 0, 512, 512, WT, (const float*)d_in[IPROJB], BB, stream);
  gemm(HA, WT, BB, nullptr, HB, 16, 512, 512, 0, stream);
  // point_features
  prep((const float*)d_in[IPF1], 512, 1024, 1024, 0, 512, 1024, WT, (const float*)d_in[IPF1B], BB, stream);
  gemm(HB, WT, BB, nullptr, HA, 16, 1024, 512, 1, stream);
  prep((const float*)d_in[IPF2], 1024, 512, 512, 0, 1024, 512, WT, (const float*)d_in[IPF2B], BB, stream);
  gemm(HA, WT, BB, nullptr, HB, 16, 512, 1024, 1, stream);
  // classifier head
  prep((const float*)d_in[IO1], 512, 256, 256, 0, 512, 256, WT, (const float*)d_in[IO1B], BB, stream);
  gemm(HB, WT, BB, nullptr, HA, 16, 256, 512, 1, stream);
  prep((const float*)d_in[IO2], 256, 128, 128, 0, 256, 128, WT, (const float*)d_in[IO2B], BB, stream);
  gemm(HA, WT, BB, nullptr, HB, 16, 128, 256, 1, stream);
  prep((const float*)d_in[IO3], 128, 49, 49, 0, 128, 64, WT, (const float*)d_in[IO3B], BB, stream);
  gemm(HB, WT, BB, HEADF, nullptr, 16, 64, 128, 0, stream);

  k_copyout<<<(BATCH * 49 + 255) / 256, 256, 0, stream>>>(HEADF, (float*)d_out);
}
